// RingDilatedAttentionSDPA_64690797412683
// MI455X (gfx1250) — compile-verified, matrix operand order checked by reading, not run on hardware
//
#include <hip/hip_runtime.h>
#include <cstdint>

// ---- problem constants (B=1, S=8192, E=1024, H=16, D=64) ----
#define S_LEN  8192
#define EMB    1024
#define NH     16
#define HD     64
#define QKV_LD 3072   // 3*EMB

typedef __bf16 bf16_t;
typedef bf16_t v16bf __attribute__((ext_vector_type(16)));
typedef float  v8f   __attribute__((ext_vector_type(8)));
typedef unsigned int u32x4 __attribute__((ext_vector_type(4)));

static __device__ __forceinline__ unsigned short f2bf(float f) {
  unsigned int u = __builtin_bit_cast(unsigned int, f);
  u += 0x7FFFu + ((u >> 16) & 1u);           // round-to-nearest-even
  return (unsigned short)(u >> 16);
}

// Load one 16x32 bf16 A/B fragment row-slice for this lane.
// Per ISA 7.12.2: lane (row/col) = lane&15, halfSel = lane>>4; the 16 bf16
// elements cover K = halfSel*8..+7 and K = 16+halfSel*8..+7 (two b128 loads).
static __device__ __forceinline__ v16bf load_frag(const unsigned short* rowK, int halfSel) {
  union U { u32x4 q[2]; v16bf v; } u;
  u.q[0] = *reinterpret_cast<const u32x4*>(rowK + halfSel * 8);
  u.q[1] = *reinterpret_cast<const u32x4*>(rowK + 16 + halfSel * 8);
  return u.v;
}

static __device__ __forceinline__ v8f wmma_bf16(v16bf a, v16bf b, v8f c) {
  return __builtin_amdgcn_wmma_f32_16x16x32_bf16(false, a, false, b, (short)0, c, false, false);
}

// ---------------- elementwise helpers ----------------
__global__ void pack_bf16_k(const float* __restrict__ src, unsigned short* __restrict__ dst, int n) {
  int i = blockIdx.x * blockDim.x + threadIdx.x;
  if (i < n) dst[i] = f2bf(src[i]);
}

__global__ void zero_f32_k(float* __restrict__ p, int n) {
  int i = blockIdx.x * blockDim.x + threadIdx.x;
  if (i < n) p[i] = 0.0f;
}

__global__ void scale_pack_k(const float* __restrict__ src, unsigned short* __restrict__ dst,
                             float s, int n) {
  int i = blockIdx.x * blockDim.x + threadIdx.x;
  if (i < n) dst[i] = f2bf(src[i] * s);
}

// Vt[h*64+d][segbase+i] = V[pos_i][h,d], pos_i = i*dil  (writes coalesced in i)
__global__ void gather_v_k(const unsigned short* __restrict__ qkvb, unsigned short* __restrict__ vt,
                           int n, int dil, int segbase) {
  int t = blockIdx.x * blockDim.x + threadIdx.x;
  if (t >= n * EMB) return;
  int i  = t % n;
  int hd = t / n;
  vt[(size_t)hd * S_LEN + segbase + i] = qkvb[(size_t)(i * dil) * QKV_LD + 2 * EMB + hd];
}

// ---------------- bf16 WMMA GEMM:  C[M,N] = A[M,K] * B[N,K]^T + bias[N] ----------------
// Columns < scaleCols are additionally multiplied by scaleVal (used to fold the
// 1/sqrt(D) attention scale into the Q section of the QKV projection).
// block = 256 threads = 8 waves; each wave -> 16x64 output strip. grid = (N/64, M/128)
template <bool STORE_BF16>
__global__ __launch_bounds__(256) void gemm_bf16_wmma(
    const unsigned short* __restrict__ A, int lda,
    const unsigned short* __restrict__ B, int ldb,
    const float* __restrict__ bias,
    void* __restrict__ Cout, int ldc, int K,
    int scaleCols, float scaleVal)
{
  const int lane    = threadIdx.x & 31;
  const int wave    = threadIdx.x >> 5;
  const int halfSel = lane >> 4;
  const int l16     = lane & 15;
  const int row0    = blockIdx.y * 128 + wave * 16;
  const int col0    = blockIdx.x * 64;

  v8f acc[4] = {};
  const unsigned short* aRow = A + (size_t)(row0 + l16) * lda;
  for (int k = 0; k < K; k += 32) {
    v16bf af = load_frag(aRow + k, halfSel);
#pragma unroll
    for (int j = 0; j < 4; ++j) {
      const unsigned short* bRow = B + (size_t)(col0 + j * 16 + l16) * ldb + k;
      v16bf bf = load_frag(bRow, halfSel);
      acc[j] = wmma_bf16(af, bf, acc[j]);
    }
  }
#pragma unroll
  for (int j = 0; j < 4; ++j) {
    const int col = col0 + j * 16 + l16;
    const float bv = bias[col];
    const float sc = (col < scaleCols) ? scaleVal : 1.0f;
#pragma unroll
    for (int r = 0; r < 8; ++r) {
      const int row = row0 + r + 8 * halfSel;   // C layout: vgpr r -> M = r + 8*(lane>>4)
      const float v = (acc[j][r] + bv) * sc;
      if (STORE_BF16) ((unsigned short*)Cout)[(size_t)row * ldc + col] = f2bf(v);
      else            ((float*)Cout)[(size_t)row * ldc + col] = v;
    }
  }
}

// ---------------- flash attention over one dilated segment (transposed form) ----------------
// Computes S^T = K * Q^T so the kv axis is in-lane: softmax reductions are
// in-register + one cross-half shuffle, and the S^T C-layout coincides exactly
// with the B-fragment K-pattern of P^T, so P repacks in registers (no LDS).
// Output accumulated as O^T = V^T * P^T; per-lane m/l scalars align with O^T cols.
// block = 8 waves; wave -> 16 queries. grid = (n/128, NH).
__global__ __launch_bounds__(256) void attn_seg_wmma(
    const unsigned short* __restrict__ qkvb,   // [S][3E] bf16 (q at +0 (pre-scaled), k at +E, v at +2E)
    const unsigned short* __restrict__ vt,     // [E rows][S_LEN] bf16, row = h*64+d
    float* __restrict__ attn,                  // [S][E] f32 accumulator (+=)
    int n, int dil, int segbase)
{
  const int lane    = threadIdx.x & 31;
  const int wave    = threadIdx.x >> 5;
  const int halfSel = lane >> 4;
  const int l16     = lane & 15;
  const int h       = blockIdx.y;
  const int q0      = (blockIdx.x * 8 + wave) * 16;

  // Q^T as the B operand: lane = query column q0+l16 (head dim = two K-chunks of 32)
  const unsigned short* qRow = qkvb + (size_t)((q0 + l16) * dil) * QKV_LD + h * HD;
  const v16bf qf0 = load_frag(qRow, halfSel);
  const v16bf qf1 = load_frag(qRow + 32, halfSel);

  v8f acc[4] = {};            // O^T tile j: rows d = 16j + r + 8*half, col q = l16
  float m1 = -3.0e38f;        // per-lane running max for query column l16
  float l1 = 0.0f;            // per-lane running denom

  for (int kb = 0; kb < n; kb += 32) {
    // ---- S^T tiles (rows kv, col q): s0 -> kv 0..15, s1 -> kv 16..31 ----
    v8f s0 = {}, s1 = {};
    {
      const unsigned short* kRow0 = qkvb + (size_t)((kb + l16) * dil) * QKV_LD + EMB + h * HD;
      const unsigned short* kRow1 = qkvb + (size_t)((kb + 16 + l16) * dil) * QKV_LD + EMB + h * HD;
      s0 = wmma_bf16(load_frag(kRow0,      halfSel), qf0, s0);
      s0 = wmma_bf16(load_frag(kRow0 + 32, halfSel), qf1, s0);
      s1 = wmma_bf16(load_frag(kRow1,      halfSel), qf0, s1);
      s1 = wmma_bf16(load_frag(kRow1 + 32, halfSel), qf1, s1);
    }
    // ---- online softmax: in-lane reduce over 16 kv values + one half-swap shuffle ----
    float t = fmaxf(s0[0], s1[0]);
#pragma unroll
    for (int r = 1; r < 8; ++r) t = fmaxf(t, fmaxf(s0[r], s1[r]));
    t = fmaxf(t, __shfl_xor(t, 16));
    const float mnew  = fmaxf(m1, t);
    const float alpha = __expf(m1 - mnew);
    m1 = mnew;

    float p0a[8], p1a[8];
    float rs = 0.0f;
#pragma unroll
    for (int r = 0; r < 8; ++r) {
      p0a[r] = __expf(s0[r] - mnew);
      p1a[r] = __expf(s1[r] - mnew);
      rs += p0a[r] + p1a[r];
    }
    rs += __shfl_xor(rs, 16);
    l1 = l1 * alpha + rs;
#pragma unroll
    for (int j = 0; j < 4; ++j)
#pragma unroll
      for (int r = 0; r < 8; ++r) acc[j][r] *= alpha;

    // ---- repack P^T B-fragment in registers: elements 0..7 = kv 8h+r, 8..15 = kv 16+8h+r ----
    union { v16bf v; unsigned int w[8]; } pu;
#pragma unroll
    for (int r = 0; r < 8; r += 2) {
      pu.w[r >> 1]       = (unsigned)f2bf(p0a[r]) | ((unsigned)f2bf(p0a[r + 1]) << 16);
      pu.w[4 + (r >> 1)] = (unsigned)f2bf(p1a[r]) | ((unsigned)f2bf(p1a[r + 1]) << 16);
    }

    // ---- O^T += V^T * P^T  (V^T rows are kv-contiguous A-fragments) ----
#pragma unroll
    for (int j = 0; j < 4; ++j) {
      const unsigned short* vRow = vt + (size_t)(h * HD + j * 16 + l16) * S_LEN + segbase + kb;
      acc[j] = wmma_bf16(load_frag(vRow, halfSel), pu.v, acc[j]);
    }
  }

  // ---- normalize (per-lane scalar) and scatter O^T into the accumulator ----
  const float inv = (l1 > 0.0f) ? 1.0f / l1 : 0.0f;
  const int token = (q0 + l16) * dil;
  float* outBase = attn + (size_t)token * EMB + h * HD;
#pragma unroll
  for (int j = 0; j < 4; ++j)
#pragma unroll
    for (int r = 0; r < 8; ++r)
      outBase[j * 16 + r + 8 * halfSel] += acc[j][r] * inv;
}

// ---------------- host driver ----------------
extern "C" void kernel_launch(void* const* d_in, const int* in_sizes, int n_in,
                              void* d_out, int out_size, void* d_ws, size_t ws_size,
                              hipStream_t stream) {
  const float* x     = (const float*)d_in[0];
  const float* w_qkv = (const float*)d_in[1];
  const float* b_qkv = (const float*)d_in[2];
  const float* w_out = (const float*)d_in[3];
  const float* b_out = (const float*)d_in[4];

  char* ws = (char*)d_ws;
  size_t off = 0;
  auto carve = [&](size_t bytes) -> char* {
    char* p = ws + off;
    off += (bytes + 255) & ~(size_t)255;
    return p;
  };
  unsigned short* xb    = (unsigned short*)carve((size_t)S_LEN * EMB * 2);
  unsigned short* wqkvb = (unsigned short*)carve((size_t)3 * EMB * EMB * 2);
  unsigned short* woutb = (unsigned short*)carve((size_t)EMB * EMB * 2);
  unsigned short* qkvb  = (unsigned short*)carve((size_t)S_LEN * QKV_LD * 2);
  unsigned short* vt    = (unsigned short*)carve((size_t)EMB * S_LEN * 2);
  float*          attn  = (float*)carve((size_t)S_LEN * EMB * 4);
  unsigned short* attnb = (unsigned short*)carve((size_t)S_LEN * EMB * 2);

  const int nx = S_LEN * EMB;
  const int nw = 3 * EMB * EMB;
  const int no = EMB * EMB;
  pack_bf16_k<<<(nx + 255) / 256, 256, 0, stream>>>(x, xb, nx);
  pack_bf16_k<<<(nw + 255) / 256, 256, 0, stream>>>(w_qkv, wqkvb, nw);
  pack_bf16_k<<<(no + 255) / 256, 256, 0, stream>>>(w_out, woutb, no);

  // QKV projection: qkv[8192,3072] = x * w_qkv^T + b_qkv (bf16 out);
  // Q columns (< EMB) pre-scaled by 1/sqrt(D) = 0.125.
  gemm_bf16_wmma<true><<<dim3(QKV_LD / 64, S_LEN / 128), 256, 0, stream>>>(
      xb, EMB, wqkvb, EMB, b_qkv, qkvb, QKV_LD, EMB, EMB, 0.125f);

  zero_f32_k<<<(nx + 255) / 256, 256, 0, stream>>>(attn, nx);

  const int segN[3]    = {2048, 4096, 2048};   // gathered positions per segment
  const int segDil[3]  = {1, 2, 4};
  const int segBase[3] = {0, 2048, 6144};      // packing offsets in vt
  for (int s = 0; s < 3; ++s) {
    const int tot = segN[s] * EMB;
    gather_v_k<<<(tot + 255) / 256, 256, 0, stream>>>(qkvb, vt, segN[s], segDil[s], segBase[s]);
  }
  for (int s = 0; s < 3; ++s) {
    attn_seg_wmma<<<dim3(segN[s] / 128, NH), 256, 0, stream>>>(
        qkvb, vt, attn, segN[s], segDil[s], segBase[s]);
  }

  // mean over 3 segments folded into the bf16 pack
  scale_pack_k<<<(nx + 255) / 256, 256, 0, stream>>>(attn, attnb, 1.0f / 3.0f, nx);

  // output projection: out[8192,1024] = attn * w_out^T + b_out  (f32 out)
  gemm_bf16_wmma<false><<<dim3(EMB / 64, S_LEN / 128), 256, 0, stream>>>(
      attnb, EMB, woutb, EMB, b_out, d_out, EMB, EMB, 0, 1.0f);
}
